// RetNet_77799037600359
// MI455X (gfx1250) — matrix operand order repaired and checked by reading, not compiled, not run on hardware
//
#include <hip/hip_runtime.h>
#include <hip/hip_bf16.h>
#include <math.h>

// Problem constants (from the reference)
#define BATCH   4
#define SEQ     4096
#define HDIM    2048
#define NHEADS  8
#define CHUNKC  256
#define DHEAD   256      // HDIM / NHEADS
#define NCHUNK  16       // SEQ / CHUNKC
#define ESLICE  64       // state-column slice per retention block

typedef __attribute__((ext_vector_type(4)))  float  v4f;
typedef __attribute__((ext_vector_type(8)))  float  v8f;
typedef __attribute__((ext_vector_type(4)))  int    v4i;
typedef __attribute__((ext_vector_type(8)))  __bf16 v8bf;
typedef __attribute__((ext_vector_type(16))) __bf16 v16bf;

__device__ inline v8f vzero() {
  v8f z;
#pragma unroll
  for (int i = 0; i < 8; ++i) z[i] = 0.0f;
  return z;
}

__device__ inline v16bf cat16(v8bf lo, v8bf hi) {
  v16bf r;
#pragma unroll
  for (int t = 0; t < 8; ++t) { r[t] = lo[t]; r[8 + t] = hi[t]; }
  return r;
}

__device__ inline v8f wmma_bf16(v16bf a, v16bf b, v8f c) {
  return __builtin_amdgcn_wmma_f32_16x16x32_bf16(
      false, a, false, b, (short)0, c, false, false);
}

// gammas: betas = linspace(log2(1/32), log2(1/512), 8); g = 1 - 2^beta
__device__ inline float head_log2g(int hh) {
  float beta = -5.0f - 4.0f * (float)hh / 7.0f;
  float g = 1.0f - exp2f(beta);
  return log2f(g);
}

// ---------------------------------------------------------------------------
// CDNA5 async global->LDS copy path (guarded; falls back to plain copies).
// ---------------------------------------------------------------------------
#if defined(__gfx1250__) && __has_builtin(__builtin_amdgcn_global_load_async_to_lds_b128)
#define USE_ASYNC_COPY 1
#else
#define USE_ASYNC_COPY 0
#endif

#if USE_ASYNC_COPY
typedef __attribute__((address_space(1))) v4i as1_v4i;
typedef __attribute__((address_space(3))) v4i as3_v4i;

__device__ __forceinline__ void async_copy16(const __bf16* gsrc, __bf16* ldst) {
  // 16-byte async copy: LDS[ldst] = MEM[gsrc]; tracked by ASYNCcnt.
  __builtin_amdgcn_global_load_async_to_lds_b128(
      (as1_v4i*)(unsigned long long)gsrc,
      (as3_v4i*)(unsigned int)(unsigned long long)ldst,
      0, 0);
}
#endif

// ---------------------------------------------------------------------------
// Kernel 0a: convert x (f32) -> xb (bf16), 8 elements per thread, vectorized.
// ---------------------------------------------------------------------------
__global__ __launch_bounds__(256) void k_cvt_x(const float* __restrict__ x,
                                               __bf16* __restrict__ xb) {
  size_t i = ((size_t)blockIdx.x * 256 + threadIdx.x) * 8;
  v4f a = *(const v4f*)(x + i);
  v4f b = *(const v4f*)(x + i + 4);
  v8bf o;
#pragma unroll
  for (int t = 0; t < 4; ++t) { o[t] = (__bf16)a[t]; o[4 + t] = (__bf16)b[t]; }
  *(v8bf*)(xb + i) = o;
}

// ---------------------------------------------------------------------------
// Kernel 0b: convert + transpose W (f32, [k][n]) -> Wt (bf16, [n][k]).
// ---------------------------------------------------------------------------
__global__ __launch_bounds__(256) void k_cvt_wt(
    const float* __restrict__ Wq, const float* __restrict__ Wk,
    const float* __restrict__ Wv,
    __bf16* __restrict__ Wtq, __bf16* __restrict__ Wtk, __bf16* __restrict__ Wtv)
{
  __shared__ __bf16 tile[64 * 65];
  const float* W; __bf16* Wt;
  if (blockIdx.z == 0)      { W = Wq; Wt = Wtq; }
  else if (blockIdx.z == 1) { W = Wk; Wt = Wtk; }
  else                      { W = Wv; Wt = Wtv; }

  const int k0 = blockIdx.y * 64;
  const int n0 = blockIdx.x * 64;
  const int tid = threadIdx.x;
  const int c = tid & 63;
  const int rb = tid >> 6;

#pragma unroll
  for (int p = 0; p < 16; ++p) {
    int r = p * 4 + rb;
    tile[c * 65 + r] = (__bf16)W[(size_t)(k0 + r) * HDIM + n0 + c];
  }
  __syncthreads();
#pragma unroll
  for (int p = 0; p < 16; ++p) {
    int r = p * 4 + rb;
    Wt[(size_t)(n0 + r) * HDIM + k0 + c] = tile[r * 65 + c];
  }
}

// ---------------------------------------------------------------------------
// Kernel 1: QKV projection GEMM on bf16 inputs, double-buffered async pipeline.
// C[16384x2048] = Xb[16384x2048] * Wt^T (Wt stored [n][k]).
// 128x128 block tile, K-step 64; per K-step each lane issues 8 async b128
// copies (4 for A, 4 for B).  While computing buf[cur] the next tile's copies
// are already in flight; s_wait_asynccnt 8 retires only the current tile.
// ---------------------------------------------------------------------------
__global__ __launch_bounds__(256) void k_qkv(
    const __bf16* __restrict__ xb,
    const __bf16* __restrict__ Wtq, const float* __restrict__ bq,
    const __bf16* __restrict__ Wtk, const float* __restrict__ bk,
    const __bf16* __restrict__ Wtv, const float* __restrict__ bv,
    __bf16* __restrict__ qb, __bf16* __restrict__ kb, __bf16* __restrict__ vb)
{
  __shared__ __bf16 sA[2][128 * 64];   // [buf][m][k]  16KB each
  __shared__ __bf16 sB[2][128 * 64];   // [buf][n][k]  16KB each

  const __bf16* Wt; const float* bias; __bf16* outp;
  if (blockIdx.z == 0)      { Wt = Wtq; bias = bq; outp = qb; }
  else if (blockIdx.z == 1) { Wt = Wtk; bias = bk; outp = kb; }
  else                      { Wt = Wtv; bias = bv; outp = vb; }

  const int m0 = blockIdx.y * 128;
  const int n0 = blockIdx.x * 128;
  const int tid = threadIdx.x;
  const int lane = tid & 31;
  const int wave = tid >> 5;
  const int wm = (wave >> 1) * 32;   // 4 wave-rows
  const int wn = (wave & 1) * 64;    // 2 wave-cols
  const int khalf = lane >> 4;
  const int l16 = lane & 15;

  v8f acc[2][4];
#pragma unroll
  for (int mt = 0; mt < 2; ++mt)
#pragma unroll
    for (int nt = 0; nt < 4; ++nt) acc[mt][nt] = vzero();

  // Stage one 128x64 A tile + 128x64 B tile into buffer `buf` for K-offset k0.
  auto stage = [&](int buf, int k0) {
#pragma unroll
    for (int p = 0; p < 4; ++p) {
      int idx = p * 256 + tid;       // [0,1024)
      int r = idx >> 3;              // row 0..127
      int q = idx & 7;               // 16B chunk within the 128B row
      const __bf16* srcA = xb + (size_t)(m0 + r) * HDIM + k0 + q * 8;
      const __bf16* srcB = Wt + (size_t)(n0 + r) * HDIM + k0 + q * 8;
      __bf16* dstA = &sA[buf][r * 64 + q * 8];
      __bf16* dstB = &sB[buf][r * 64 + q * 8];
#if USE_ASYNC_COPY
      async_copy16(srcA, dstA);
      async_copy16(srcB, dstB);
#else
      *(v8bf*)dstA = *(const v8bf*)srcA;
      *(v8bf*)dstB = *(const v8bf*)srcB;
#endif
    }
  };

  stage(0, 0);
  for (int k0 = 0; k0 < HDIM; k0 += 64) {
    const int cur = (k0 >> 6) & 1;
    const bool more = (k0 + 64) < HDIM;
    if (more) stage(1 - cur, k0 + 64);   // next tile's copies go in flight
#if USE_ASYNC_COPY
    if (more) __builtin_amdgcn_s_wait_asynccnt(8);   // current tile retired
    else      __builtin_amdgcn_s_wait_asynccnt(0);
#endif
    __syncthreads();

#pragma unroll
    for (int ks = 0; ks < 64; ks += 32) {
      v16bf afr[2], bfr[4];
#pragma unroll
      for (int mt = 0; mt < 2; ++mt) {
        const __bf16* p = &sA[cur][(wm + mt * 16 + l16) * 64 + ks + khalf * 8];
        afr[mt] = cat16(*(const v8bf*)p, *(const v8bf*)(p + 16));
      }
#pragma unroll
      for (int nt = 0; nt < 4; ++nt) {
        const __bf16* p = &sB[cur][(wn + nt * 16 + l16) * 64 + ks + khalf * 8];
        bfr[nt] = cat16(*(const v8bf*)p, *(const v8bf*)(p + 16));
      }
#pragma unroll
      for (int mt = 0; mt < 2; ++mt)
#pragma unroll
        for (int nt = 0; nt < 4; ++nt)
          acc[mt][nt] = wmma_bf16(afr[mt], bfr[nt], acc[mt][nt]);
    }
    __syncthreads();   // all waves done reading buf[cur] before it is restaged
  }

#pragma unroll
  for (int mt = 0; mt < 2; ++mt)
#pragma unroll
    for (int nt = 0; nt < 4; ++nt)
#pragma unroll
      for (int r = 0; r < 8; ++r) {
        int m = m0 + wm + mt * 16 + r + 8 * khalf;
        int n = n0 + wn + nt * 16 + l16;
        float vv = acc[mt][nt][r] + bias[n];
        outp[(size_t)m * HDIM + n] = (__bf16)vv;
      }
}

// ---------------------------------------------------------------------------
// Kernel 2: masked intra-chunk attention.  attn = (q_chunk @ k_chunk^T) * D
// ---------------------------------------------------------------------------
__global__ __launch_bounds__(256) void k_attn(
    const __bf16* __restrict__ qb, const __bf16* __restrict__ kb,
    __bf16* __restrict__ attn)
{
  const int z = blockIdx.z;               // ((bb*8 + hh)*16 + n)
  const int n  = z & 15;
  const int hh = (z >> 4) & 7;
  const int bb = z >> 7;
  const float lg = head_log2g(hh);

  const __bf16* qc = qb + ((size_t)bb * SEQ + (size_t)n * CHUNKC) * HDIM + hh * DHEAD;
  const __bf16* kc = kb + ((size_t)bb * SEQ + (size_t)n * CHUNKC) * HDIM + hh * DHEAD;
  __bf16* ac = attn + (size_t)z * CHUNKC * CHUNKC;

  const int i0 = blockIdx.y * 128;
  const int j0 = blockIdx.x * 128;
  const int tid = threadIdx.x;
  const int lane = tid & 31;
  const int wave = tid >> 5;
  const int wm = (wave >> 1) * 32;
  const int wn = (wave & 1) * 64;
  const int khalf = lane >> 4;
  const int l16 = lane & 15;

  v8f acc[2][4];
#pragma unroll
  for (int mt = 0; mt < 2; ++mt)
#pragma unroll
    for (int nt = 0; nt < 4; ++nt) acc[mt][nt] = vzero();

  for (int k0 = 0; k0 < DHEAD; k0 += 32) {
    v16bf afr[2], bfr[4];
#pragma unroll
    for (int mt = 0; mt < 2; ++mt) {
      const __bf16* p = qc + (size_t)(i0 + wm + mt * 16 + l16) * HDIM + k0 + khalf * 8;
      afr[mt] = cat16(*(const v8bf*)p, *(const v8bf*)(p + 16));
    }
#pragma unroll
    for (int nt = 0; nt < 4; ++nt) {
      const __bf16* p = kc + (size_t)(j0 + wn + nt * 16 + l16) * HDIM + k0 + khalf * 8;
      bfr[nt] = cat16(*(const v8bf*)p, *(const v8bf*)(p + 16));
    }
#pragma unroll
    for (int mt = 0; mt < 2; ++mt)
#pragma unroll
      for (int nt = 0; nt < 4; ++nt)
        acc[mt][nt] = wmma_bf16(afr[mt], bfr[nt], acc[mt][nt]);
  }

#pragma unroll
  for (int mt = 0; mt < 2; ++mt)
#pragma unroll
    for (int nt = 0; nt < 4; ++nt)
#pragma unroll
      for (int r = 0; r < 8; ++r) {
        int i = i0 + wm + mt * 16 + r + 8 * khalf;
        int j = j0 + wn + nt * 16 + l16;
        float val = (i >= j) ? acc[mt][nt][r] * exp2f((float)(i - j) * lg) : 0.0f;
        ac[(size_t)i * CHUNKC + j] = (__bf16)val;
      }
}

// ---------------------------------------------------------------------------
// Kernel 3: retention recurrence.  One block per (b, h, 64-col state slice).
// LDS: f32 state Sst[d][e], bf16 shadow SbfT[e][d], vT[e][c], kTs[d][c'].
// All WMMA fragments are contiguous ds_load_b128 pairs.
// ---------------------------------------------------------------------------
__global__ __launch_bounds__(256) void k_ret(
    const __bf16* __restrict__ qb, const __bf16* __restrict__ kb,
    const __bf16* __restrict__ vb, const __bf16* __restrict__ attn,
    float* __restrict__ out)
{
  __shared__ float  Sst[DHEAD * ESLICE];    // [dd][e]
  __shared__ __bf16 SbfT[ESLICE * DHEAD];   // [e][dd]
  __shared__ __bf16 vT[ESLICE * CHUNKC];    // [e][c]
  __shared__ __bf16 kTs[DHEAD * 32];        // [dd][c-k0], kv-decay folded in
  __shared__ float  scKV[CHUNKC];           // g^(C-1-c)

  const int e0 = blockIdx.x * ESLICE;
  const int hh = blockIdx.y;
  const int bb = blockIdx.z;
  const float lg = head_log2g(hh);
  const float cdecay = exp2f((float)CHUNKC * lg);   // g^C

  const int tid = threadIdx.x;
  const int lane = tid & 31;
  const int wave = tid >> 5;
  const int wm = wave * 32;
  const int khalf = lane >> 4;
  const int l16 = lane & 15;

  for (int i = tid; i < DHEAD * ESLICE; i += 256) Sst[i] = 0.0f;
  for (int i = tid; i < ESLICE * DHEAD; i += 256) SbfT[i] = (__bf16)0.0f;
  scKV[tid] = exp2f((float)(CHUNKC - 1 - tid) * lg);
  __syncthreads();

  float csc[2];
#pragma unroll
  for (int mt = 0; mt < 2; ++mt)
    csc[mt] = exp2f((float)(wm + mt * 16 + l16 + 1) * lg);

  for (int n = 0; n < NCHUNK; ++n) {
    const __bf16* qc = qb + ((size_t)bb * SEQ + (size_t)n * CHUNKC) * HDIM + hh * DHEAD;
    const __bf16* kc = kb + ((size_t)bb * SEQ + (size_t)n * CHUNKC) * HDIM + hh * DHEAD;
    const __bf16* vc = vb + ((size_t)bb * SEQ + (size_t)n * CHUNKC) * HDIM + hh * DHEAD + e0;
    const __bf16* ac = attn + (((size_t)bb * NHEADS + hh) * NCHUNK + n) * CHUNKC * CHUNKC;

    // stage vT[e][c] (coalesced global reads of v rows)
    {
      int col = tid & 63;
      int rb = tid >> 6;
#pragma unroll 4
      for (int p = 0; p < 64; ++p) {
        int c = p * 4 + rb;
        vT[col * CHUNKC + c] = vc[(size_t)c * HDIM + col];
      }
    }
    __syncthreads();

    // ---- Phase A: out = attn @ v_slice + (q * g^{i+1}) @ S ----
    v8f acc[2][4];
#pragma unroll
    for (int mt = 0; mt < 2; ++mt)
#pragma unroll
      for (int nt = 0; nt < 4; ++nt) acc[mt][nt] = vzero();

    for (int k0 = 0; k0 < CHUNKC; k0 += 32) {       // inner: attn @ v
      v16bf afr[2], bfr[4];
#pragma unroll
      for (int mt = 0; mt < 2; ++mt) {
        const __bf16* p = ac + (size_t)(wm + mt * 16 + l16) * CHUNKC + k0 + khalf * 8;
        afr[mt] = cat16(*(const v8bf*)p, *(const v8bf*)(p + 16));
      }
#pragma unroll
      for (int nt = 0; nt < 4; ++nt) {
        const __bf16* p = &vT[(nt * 16 + l16) * CHUNKC + k0 + khalf * 8];
        bfr[nt] = cat16(*(const v8bf*)p, *(const v8bf*)(p + 16));
      }
#pragma unroll
      for (int mt = 0; mt < 2; ++mt)
#pragma unroll
        for (int nt = 0; nt < 4; ++nt)
          acc[mt][nt] = wmma_bf16(afr[mt], bfr[nt], acc[mt][nt]);
    }

    for (int k0 = 0; k0 < DHEAD; k0 += 32) {        // cross: q_scaled @ S
      v16bf afr[2], bfr[4];
#pragma unroll
      for (int mt = 0; mt < 2; ++mt) {
        const __bf16* p = qc + (size_t)(wm + mt * 16 + l16) * HDIM + k0 + khalf * 8;
        v8bf lo = *(const v8bf*)p;
        v8bf hi = *(const v8bf*)(p + 16);
        v16bf a;
#pragma unroll
        for (int t = 0; t < 8; ++t) {
          a[t]     = (__bf16)((float)lo[t] * csc[mt]);
          a[8 + t] = (__bf16)((float)hi[t] * csc[mt]);
        }
        afr[mt] = a;
      }
#pragma unroll
      for (int nt = 0; nt < 4; ++nt) {
        const __bf16* p = &SbfT[(nt * 16 + l16) * DHEAD + k0 + khalf * 8];
        bfr[nt] = cat16(*(const v8bf*)p, *(const v8bf*)(p + 16));
      }
#pragma unroll
      for (int mt = 0; mt < 2; ++mt)
#pragma unroll
        for (int nt = 0; nt < 4; ++nt)
          acc[mt][nt] = wmma_bf16(afr[mt], bfr[nt], acc[mt][nt]);
    }

    // store output (f32)
#pragma unroll
    for (int mt = 0; mt < 2; ++mt)
#pragma unroll
      for (int nt = 0; nt < 4; ++nt)
#pragma unroll
        for (int r = 0; r < 8; ++r) {
          int i = wm + mt * 16 + r + 8 * khalf;
          int col = nt * 16 + l16;
          out[((size_t)bb * SEQ + (size_t)n * CHUNKC + i) * HDIM + hh * DHEAD + e0 + col]
              = acc[mt][nt][r];
        }
    __syncthreads();   // all reads of SbfT for this chunk are done

    // ---- Phase B: T = (k * g^{C-1-c})^T @ v_slice ----
    v8f accT[2][4];
#pragma unroll
    for (int mt = 0; mt < 2; ++mt)
#pragma unroll
      for (int nt = 0; nt < 4; ++nt) accT[mt][nt] = vzero();

    for (int k0 = 0; k0 < CHUNKC; k0 += 32) {
#pragma unroll 8
      for (int p = 0; p < 32; ++p) {
        float sc = scKV[k0 + p];
        kTs[tid * 32 + p] = (__bf16)((float)kc[(size_t)(k0 + p) * HDIM + tid] * sc);
      }
      __syncthreads();

      v16bf afr[2], bfr[4];
#pragma unroll
      for (int mt = 0; mt < 2; ++mt) {
        const __bf16* p = &kTs[(wm + mt * 16 + l16) * 32 + khalf * 8];
        afr[mt] = cat16(*(const v8bf*)p, *(const v8bf*)(p + 16));
      }
#pragma unroll
      for (int nt = 0; nt < 4; ++nt) {
        const __bf16* p = &vT[(nt * 16 + l16) * CHUNKC + k0 + khalf * 8];
        bfr[nt] = cat16(*(const v8bf*)p, *(const v8bf*)(p + 16));
      }
#pragma unroll
      for (int mt = 0; mt < 2; ++mt)
#pragma unroll
        for (int nt = 0; nt < 4; ++nt)
          accT[mt][nt] = wmma_bf16(afr[mt], bfr[nt], accT[mt][nt]);
      __syncthreads();   // before kTs is overwritten next step
    }

    // S = g^C * S + T ; refresh bf16 shadow
#pragma unroll
    for (int mt = 0; mt < 2; ++mt)
#pragma unroll
      for (int nt = 0; nt < 4; ++nt)
#pragma unroll
        for (int r = 0; r < 8; ++r) {
          int dd = wm + mt * 16 + r + 8 * khalf;
          int col = nt * 16 + l16;
          float nv = cdecay * Sst[dd * ESLICE + col] + accT[mt][nt][r];
          Sst[dd * ESLICE + col] = nv;
          SbfT[col * DHEAD + dd] = (__bf16)nv;
        }
    __syncthreads();   // state ready for next chunk
  }
}

// ---------------------------------------------------------------------------
// ws layout (bf16 elements):
//   xb (33,554,432) | Wtq,Wtk,Wtv (3 x 4,194,304) | qb,kb,vb (3 x 33,554,432)
//   | attn (33,554,432)   -> ~344 MB total
// ---------------------------------------------------------------------------
extern "C" void kernel_launch(void* const* d_in, const int* in_sizes, int n_in,
                              void* d_out, int out_size, void* d_ws, size_t ws_size,
                              hipStream_t stream) {
  const float* x  = (const float*)d_in[0];
  const float* Wq = (const float*)d_in[1];
  const float* bq = (const float*)d_in[2];
  const float* Wk = (const float*)d_in[3];
  const float* bk = (const float*)d_in[4];
  const float* Wv = (const float*)d_in[5];
  const float* bv = (const float*)d_in[6];
  float* out = (float*)d_out;

  const size_t elems  = (size_t)BATCH * SEQ * HDIM;   // 33,554,432
  const size_t welems = (size_t)HDIM * HDIM;          //  4,194,304
  __bf16* xb   = (__bf16*)d_ws;
  __bf16* Wtq  = xb + elems;
  __bf16* Wtk  = Wtq + welems;
  __bf16* Wtv  = Wtk + welems;
  __bf16* qb   = Wtv + welems;
  __bf16* kb   = qb + elems;
  __bf16* vb   = kb + elems;
  __bf16* attn = vb + elems;

  k_cvt_x<<<dim3((unsigned)(elems / (256 * 8))), 256, 0, stream>>>(x, xb);
  k_cvt_wt<<<dim3(32, 32, 3), 256, 0, stream>>>(Wq, Wk, Wv, Wtq, Wtk, Wtv);
  k_qkv<<<dim3(HDIM / 128, (BATCH * SEQ) / 128, 3), 256, 0, stream>>>(
      xb, Wtq, bq, Wtk, bk, Wtv, bv, qb, kb, vb);
  k_attn<<<dim3(2, 2, BATCH * NHEADS * NCHUNK), 256, 0, stream>>>(qb, kb, attn);
  k_ret<<<dim3(DHEAD / ESLICE, NHEADS, BATCH), 256, 0, stream>>>(qb, kb, vb, attn, out);
}
